// BCELoss_64330020159675
// MI455X (gfx1250) — matrix operand order repaired and checked by reading, not compiled, not run on hardware
//
#include <hip/hip_runtime.h>
#include <hip/hip_bf16.h>
#include <math.h>

// Problem constants (match reference)
#define BN 8192   // batch
#define DD 1024   // embedding dim (K)
#define CC 1024   // classes (N)

typedef _Float16 h8  __attribute__((ext_vector_type(8)));
typedef _Float16 v16h __attribute__((ext_vector_type(16)));
typedef float    v8f  __attribute__((ext_vector_type(8)));
typedef int      v4i  __attribute__((ext_vector_type(4)));

#if __has_builtin(__builtin_amdgcn_global_load_async_to_lds_b128) && \
    __has_builtin(__builtin_amdgcn_s_wait_asynccnt)
#define USE_ASYNC 1
#else
#define USE_ASYNC 0
#endif

#define AS1 __attribute__((address_space(1)))
#define AS3 __attribute__((address_space(3)))

// ---------------------------------------------------------------------------
// Kernel 0: zero prototype accumulators, counts, and the output scalar.
// ---------------------------------------------------------------------------
__global__ void k_zero(float* __restrict__ protoSum, float* __restrict__ counts,
                       float* __restrict__ out) {
    int i = blockIdx.x * blockDim.x + threadIdx.x;
    if (i < CC * DD) protoSum[i] = 0.0f;
    if (i < CC)      counts[i]   = 0.0f;
    if (i == 0)      out[0]      = 0.0f;
}

// ---------------------------------------------------------------------------
// Kernel 1: normalize emb_j rows -> f16 z_j, and zjSq[row] = ||z_j_row||^2.
// ---------------------------------------------------------------------------
__global__ void k_norm_j(const float* __restrict__ emb,
                         _Float16* __restrict__ zjh,
                         float* __restrict__ zjSq) {
    const int row = blockIdx.x;
    const float* x = emb + (size_t)row * DD;
    __shared__ float red[256];
    float s = 0.0f;
    for (int d = threadIdx.x; d < DD; d += 256) { float v = x[d]; s += v * v; }
    red[threadIdx.x] = s; __syncthreads();
    for (int st = 128; st > 0; st >>= 1) {
        if (threadIdx.x < st) red[threadIdx.x] += red[threadIdx.x + st];
        __syncthreads();
    }
    const float n2  = red[0];
    const float inv = 1.0f / fmaxf(sqrtf(n2), 1e-12f);
    _Float16* dst = zjh + (size_t)row * DD;
    for (int d = threadIdx.x; d < DD; d += 256) dst[d] = (_Float16)(x[d] * inv);
    if (threadIdx.x == 0) zjSq[row] = n2 * inv * inv;
}

// ---------------------------------------------------------------------------
// Kernel 2: normalize emb_i rows, atomic-scatter into protoSum[label] + counts.
// ---------------------------------------------------------------------------
__global__ void k_acc_proto(const float* __restrict__ emb,
                            const int* __restrict__ labels,
                            float* __restrict__ protoSum,
                            float* __restrict__ counts) {
    const int row = blockIdx.x;
    const float* x = emb + (size_t)row * DD;
    __shared__ float red[256];
    float s = 0.0f;
    for (int d = threadIdx.x; d < DD; d += 256) { float v = x[d]; s += v * v; }
    red[threadIdx.x] = s; __syncthreads();
    for (int st = 128; st > 0; st >>= 1) {
        if (threadIdx.x < st) red[threadIdx.x] += red[threadIdx.x + st];
        __syncthreads();
    }
    const float inv = 1.0f / fmaxf(sqrtf(red[0]), 1e-12f);
    const int lab = labels[row];
    float* dst = protoSum + (size_t)lab * DD;
    for (int d = threadIdx.x; d < DD; d += 256) atomicAdd(dst + d, x[d] * inv);
    if (threadIdx.x == 0) atomicAdd(counts + lab, 1.0f);
}

// ---------------------------------------------------------------------------
// Kernel 3: prototypes = protoSum / count, emit f16 + per-class squared norm.
// ---------------------------------------------------------------------------
__global__ void k_fin_proto(const float* __restrict__ protoSum,
                            const float* __restrict__ counts,
                            _Float16* __restrict__ protoH,
                            float* __restrict__ protoSq) {
    const int c = blockIdx.x;
    const float inv = 1.0f / counts[c];
    const float* src = protoSum + (size_t)c * DD;
    _Float16*   dst = protoH  + (size_t)c * DD;
    __shared__ float red[256];
    float s = 0.0f;
    for (int d = threadIdx.x; d < DD; d += 256) {
        float v = src[d] * inv;
        dst[d] = (_Float16)v;
        s += v * v;
    }
    red[threadIdx.x] = s; __syncthreads();
    for (int st = 128; st > 0; st >>= 1) {
        if (threadIdx.x < st) red[threadIdx.x] += red[threadIdx.x + st];
        __syncthreads();
    }
    if (threadIdx.x == 0) protoSq[c] = red[0];
}

// ---------------------------------------------------------------------------
// Kernel 4: LDS-staged WMMA GEMM (z_j [B,K] x protoH [N,K]^T) fused with loss.
// Block (8 waves) computes 128x64 outputs; per K-step(64) the block stages
// A(128x64 f16) and B(64x64 f16) into LDS via global_load_async_to_lds_b128
// (ASYNCcnt path), then each wave runs 8x v_wmma_f32_16x16x32_f16 off LDS.
// LDS rows padded to 72 halfs (36*r mod 64 hits all bank groups -> no
// ds_load_b128 conflicts).
// Fragment packing per CDNA5 ISA 7.12.2:
//   A (16x32 f16): lane(0..15)=row M, K = {hi*8..+8, 16+hi*8..+8}
//   B (32x16 f16): lane(0..15)=col N, K = {hi*8..+8, 16+hi*8..+8} (hi = lane>>4)
//   C/D (16x16 f32): VGPR r -> M = hi*8+r, N = lane&15
// ---------------------------------------------------------------------------
#define LDP 72   // padded LDS row stride in halfs

__global__ void __launch_bounds__(256)
k_gemm_loss(const _Float16* __restrict__ zjh,
            const _Float16* __restrict__ protoH,
            const float* __restrict__ zjSq,
            const float* __restrict__ protoSq,
            const int* __restrict__ labels,
            float* __restrict__ out) {
    __shared__ _Float16 Abuf[128 * LDP];   // 18 KB
    __shared__ _Float16 Bbuf[64 * LDP];    //  9 KB

    const int wave = threadIdx.x >> 5;
    const int lane = threadIdx.x & 31;
    const int laneLo = lane & 15;
    const int hi = lane >> 4;

    // 64 M-blocks x 16 N-blocks = 1024 blocks
    const int m0 = (blockIdx.x >> 4) * 128;
    const int n0 = (blockIdx.x & 15) * 64;
    const int mw = wave * 16;              // wave's row sub-tile within block

    v8f acc[4];
#pragma unroll
    for (int t = 0; t < 4; ++t) acc[t] = (v8f){0.f,0.f,0.f,0.f,0.f,0.f,0.f,0.f};

    for (int k = 0; k < DD; k += 64) {
        // ---- stage A: 128 rows x 64 halfs = 1024 x 16B chunks, 4 per thread
#pragma unroll
        for (int i = 0; i < 4; ++i) {
            const int c = threadIdx.x + i * 256;
            const int row = c >> 3, kc = (c & 7) * 8;
            const _Float16* g = zjh + (size_t)(m0 + row) * DD + k + kc;
            _Float16* l = &Abuf[row * LDP + kc];
#if USE_ASYNC
            __builtin_amdgcn_global_load_async_to_lds_b128(
                (AS1 v4i*)g, (AS3 v4i*)l, 0, 0);
#else
            *(h8*)l = *(const h8*)g;
#endif
        }
        // ---- stage B: 64 rows x 64 halfs = 512 x 16B chunks, 2 per thread
#pragma unroll
        for (int i = 0; i < 2; ++i) {
            const int c = threadIdx.x + i * 256;
            const int row = c >> 3, kc = (c & 7) * 8;
            const _Float16* g = protoH + (size_t)(n0 + row) * DD + k + kc;
            _Float16* l = &Bbuf[row * LDP + kc];
#if USE_ASYNC
            __builtin_amdgcn_global_load_async_to_lds_b128(
                (AS1 v4i*)g, (AS3 v4i*)l, 0, 0);
#else
            *(h8*)l = *(const h8*)g;
#endif
        }
#if USE_ASYNC
        __builtin_amdgcn_s_wait_asynccnt(0);
#endif
        __syncthreads();

        // ---- compute: 2 kk-steps x 4 n-tiles = 8 WMMA per K-step
#pragma unroll
        for (int kk = 0; kk < 64; kk += 32) {
            const _Float16* al = &Abuf[(mw + laneLo) * LDP + kk + hi * 8];
            h8 alo = *(const h8*)(al);
            h8 ahi = *(const h8*)(al + 16);
            v16h afrag = __builtin_shufflevector(alo, ahi,
                            0,1,2,3,4,5,6,7,8,9,10,11,12,13,14,15);
#pragma unroll
            for (int t = 0; t < 4; ++t) {
                const _Float16* bl = &Bbuf[(t * 16 + laneLo) * LDP + kk + hi * 8];
                h8 blo = *(const h8*)(bl);
                h8 bhi = *(const h8*)(bl + 16);
                v16h bfrag = __builtin_shufflevector(blo, bhi,
                                0,1,2,3,4,5,6,7,8,9,10,11,12,13,14,15);
                acc[t] = __builtin_amdgcn_wmma_f32_16x16x32_f16(
                    false, afrag, false, bfrag, (short)0, acc[t], false, false);
            }
        }
        __syncthreads();
    }

    // ---- epilogue: d2 = |zj|^2 + |p|^2 - 2 s; sim = 2 - sqrt(max(d2,0));
    // BCE-with-logits via stable log-sigmoid; wave reduce; 1 atomic per wave.
    float pq[4];
#pragma unroll
    for (int t = 0; t < 4; ++t) pq[t] = protoSq[n0 + t * 16 + laneLo];

    float local = 0.0f;
#pragma unroll
    for (int r = 0; r < 8; ++r) {
        const int row = m0 + mw + hi * 8 + r;
        const float zq = zjSq[row];
        const int lab = labels[row];
#pragma unroll
        for (int t = 0; t < 4; ++t) {
            const int col = n0 + t * 16 + laneLo;
            const float s = acc[t][r];
            const float d2 = zq + pq[t] - 2.0f * s;
            const float sim = 2.0f - sqrtf(fmaxf(d2, 0.0f));
            const float x = (lab == col) ? sim : -sim;
            // -log_sigmoid(x) = log1p(exp(-|x|)) - min(x, 0)
            local += log1pf(expf(-fabsf(x))) - fminf(x, 0.0f);
        }
    }

#pragma unroll
    for (int off = 16; off > 0; off >>= 1)
        local += __shfl_down(local, off, 32);
    if (lane == 0)
        atomicAdd(out, local * (1.0f / ((float)BN * (float)CC)));
}

// ---------------------------------------------------------------------------
extern "C" void kernel_launch(void* const* d_in, const int* in_sizes, int n_in,
                              void* d_out, int out_size, void* d_ws, size_t ws_size,
                              hipStream_t stream) {
    (void)in_sizes; (void)n_in; (void)out_size; (void)ws_size;
    const float* emb_i  = (const float*)d_in[0];
    const float* emb_j  = (const float*)d_in[1];
    const int*   labels = (const int*)d_in[2];
    float* out = (float*)d_out;

    char* ws = (char*)d_ws;
    _Float16* zjh      = (_Float16*)(ws);                            // 16 MB
    float*    protoSum = (float*)(ws + (size_t)16 * 1024 * 1024);    //  4 MB
    _Float16* protoH   = (_Float16*)(ws + (size_t)20 * 1024 * 1024); //  2 MB
    float*    counts   = (float*)(ws + (size_t)22 * 1024 * 1024);    //  4 KB
    float*    zjSq     = (float*)(ws + (size_t)22 * 1024 * 1024 + 4096);          // 32 KB
    float*    protoSq  = (float*)(ws + (size_t)22 * 1024 * 1024 + 4096 + 32768);  //  4 KB

    k_zero<<<(CC * DD + 255) / 256, 256, 0, stream>>>(protoSum, counts, out);
    k_norm_j<<<BN, 256, 0, stream>>>(emb_j, zjh, zjSq);
    k_acc_proto<<<BN, 256, 0, stream>>>(emb_i, labels, protoSum, counts);
    k_fin_proto<<<CC, 256, 0, stream>>>(protoSum, counts, protoH, protoSq);
    k_gemm_loss<<<1024, 256, 0, stream>>>(zjh, protoH, zjSq, protoSq, labels, out);
}
